// AudioResNetFusion_47545287967360
// MI455X (gfx1250) — matrix operand order repaired and checked by reading, not compiled, not run on hardware
//
#include <hip/hip_runtime.h>
#include <hip/hip_bf16.h>
#include <math.h>

typedef __attribute__((ext_vector_type(16))) _Float16 v16h;
typedef __attribute__((ext_vector_type(8)))  _Float16 v8h;
typedef __attribute__((ext_vector_type(8)))  float    v8f;
typedef __attribute__((ext_vector_type(4)))  int      v4i;

// ---------------------------------------------------------------------------
// gfx1250 async global->LDS copy (ASYNCcnt) with compile-safe fallback
// Builtin signature (from hipcc diagnostic): (global v4i*, local v4i*, Ii, Ii)
// ---------------------------------------------------------------------------
#if defined(__AMDGCN__) && __has_builtin(__builtin_amdgcn_global_load_async_to_lds_b128)
#define ASYNC_LDS 1
#endif

__device__ __forceinline__ void cp16_to_lds(const _Float16* g, _Float16* l) {
#ifdef ASYNC_LDS
  __builtin_amdgcn_global_load_async_to_lds_b128(
      (__attribute__((address_space(1))) v4i*)g,
      (__attribute__((address_space(3))) v4i*)l, 0, 0);
#else
  *(uint4*)l = *(const uint4*)g;
#endif
}

__device__ __forceinline__ void async_wait_all() {
#ifdef ASYNC_LDS
#if __has_builtin(__builtin_amdgcn_s_wait_asynccnt)
  __builtin_amdgcn_s_wait_asynccnt(0);
#else
  asm volatile("s_wait_asynccnt 0" ::: "memory");
#endif
#endif
}

// ---------------------------------------------------------------------------
// PCEN (per-(b,f) temporal IIR) + input BN, writes f32 spec [B,128,401]
// ---------------------------------------------------------------------------
__global__ void pcen_kernel(const float* __restrict__ mel,
                            const float* ps, const float* pa, const float* pd, const float* pr,
                            const float* ig, const float* ib, const float* im, const float* iv,
                            float* __restrict__ spec, int BF, int T) {
  int idx = blockIdx.x * blockDim.x + threadIdx.x;
  if (idx >= BF) return;
  float s     = fminf(fmaxf(ps[0], 1e-4f), 1.0f);
  float alpha = fminf(fmaxf(pa[0], 0.0f), 1.0f);
  float delta = fminf(fmaxf(pd[0], 0.0f), 50.0f);
  float r     = fminf(fmaxf(pr[0], 0.0f), 1.0f);
  float dr    = powf(delta, r);
  float sc    = ig[0] * rsqrtf(iv[0] + 1e-5f);
  float sh    = ib[0] - im[0] * sc;
  const float* row = mel + (size_t)idx * T;
  float* orow = spec + (size_t)idx * T;
  float M = row[0];
  for (int t = 0; t < T; t++) {
    if (t > 0) M = (1.0f - s) * M + s * row[t];
    float sm = powf(M + 1e-6f, alpha);
    float pc = powf(row[t] / sm + delta, r) - dr;
    orow[t] = pc * sc + sh;
  }
}

// ---------------------------------------------------------------------------
// conv1 7x7 s2 p3, Cin=1 (direct; K too small for WMMA), BN+ReLU, out NHWC f16
// ---------------------------------------------------------------------------
__global__ __launch_bounds__(256) void conv1_kernel(
    const float* __restrict__ spec, const float* __restrict__ w,
    const float* g, const float* b, const float* m, const float* v,
    _Float16* __restrict__ out, int Bn, int Hin, int Win, int Ho, int Wo) {
  __shared__ float ws[64 * 49];
  for (int i = threadIdx.x; i < 64 * 49; i += blockDim.x) ws[i] = w[i];
  __syncthreads();
  int idx = blockIdx.x * blockDim.x + threadIdx.x;
  int M = Bn * Ho * Wo;
  if (idx >= M) return;
  int wo = idx % Wo; int t = idx / Wo; int ho = t % Ho; int bb = t / Ho;
  float xv[49];
  #pragma unroll
  for (int kh = 0; kh < 7; kh++)
    #pragma unroll
    for (int kw = 0; kw < 7; kw++) {
      int hin = ho * 2 - 3 + kh, win = wo * 2 - 3 + kw;
      xv[kh * 7 + kw] = (hin >= 0 && hin < Hin && win >= 0 && win < Win)
                        ? spec[((size_t)bb * Hin + hin) * Win + win] : 0.0f;
    }
  _Float16* op = out + (size_t)idx * 64;
  for (int co = 0; co < 64; co++) {
    float a = 0.0f;
    #pragma unroll
    for (int tt = 0; tt < 49; tt++) a += xv[tt] * ws[co * 49 + tt];
    float sc = g[co] * rsqrtf(v[co] + 1e-5f);
    float sh = b[co] - m[co] * sc;
    op[co] = (_Float16)fmaxf(a * sc + sh, 0.0f);
  }
}

// ---------------------------------------------------------------------------
// maxpool 3x3 s2 p1 on NHWC f16
// ---------------------------------------------------------------------------
__global__ void maxpool_kernel(const _Float16* __restrict__ X, _Float16* __restrict__ Y,
                               int Bn, int Hin, int Win, int Ho, int Wo, int C) {
  int idx = blockIdx.x * blockDim.x + threadIdx.x;
  int tot = Bn * Ho * Wo * C;
  if (idx >= tot) return;
  int c = idx % C; int p = idx / C;
  int wo = p % Wo; p /= Wo; int ho = p % Ho; int bb = p / Ho;
  float mx = -1e30f;
  for (int kh = 0; kh < 3; kh++)
    for (int kw = 0; kw < 3; kw++) {
      int hin = ho * 2 - 1 + kh, win = wo * 2 - 1 + kw;
      if (hin >= 0 && hin < Hin && win >= 0 && win < Win)
        mx = fmaxf(mx, (float)X[((size_t)(bb * Hin + hin) * Win + win) * C + c]);
    }
  Y[idx] = (_Float16)mx;
}

// ---------------------------------------------------------------------------
// weight repack: OIHW f32 -> chunk-transposed f16 layout
//   dst[(((k/32) * (Cout/64) + n/64) * 64 + n%64) * 32 + k%32]
// where k = (kh*Kw+kw)*Cin + ci, n = o.  This makes the per-(kChunk,N0) LDS
// B-image a single contiguous 4KB region -> straight async copy, no shuffles.
// ---------------------------------------------------------------------------
__global__ void repack_w_kernel(const float* __restrict__ src, _Float16* __restrict__ dst,
                                int Cout, int Cin, int K, int total) {
  int d = blockIdx.x * blockDim.x + threadIdx.x;
  if (d >= total) return;
  int kl = d & 31; int r = d >> 5;
  int nl = r & 63; r >>= 6;
  int nblocks = Cout >> 6;
  int n0b = r % nblocks; int c0 = r / nblocks;
  int k = (c0 << 5) + kl;
  int o = (n0b << 6) + nl;
  int ci = k % Cin; int tap = k / Cin;
  int kh = tap / K; int kw = tap - kh * K;
  dst[d] = (_Float16)src[((size_t)(o * Cin + ci) * K + kh) * K + kw];
}

// ---------------------------------------------------------------------------
// Implicit-GEMM conv via WMMA f16 -> f32, NHWC activations.
// Block: 256 thr = 8 waves. Block tile M=64 x N=64; wave tile 16x32
// (2 accumulators, A-fragment reused) -> 2 v_wmma per K-chunk.
// Double-buffered LDS; A and B tiles staged with GLOBAL_LOAD_ASYNC_TO_LDS_B128
// (ASYNCcnt) when available.  A is stored in natural k order (lane half hi
// takes k = 16*hi + s, same as B) so lanes fetch v16h fragments as two ds_load_b128.
// Fused BN (+residual)(+ReLU) epilogue per the documented C-layout.
// Requires Cin%32==0, Cout%64==0 (true for all ResNet stages here).
// ---------------------------------------------------------------------------
union F16Frag { v16h v; v8h h[2]; };

__global__ __launch_bounds__(256) void conv_wmma_kernel(
    const _Float16* __restrict__ X, const _Float16* __restrict__ Wt,
    const float* __restrict__ bng, const float* __restrict__ bnb,
    const float* __restrict__ bnm, const float* __restrict__ bnv,
    const _Float16* __restrict__ Res, _Float16* __restrict__ Y,
    int Bn, int Hin, int Win, int Cin, int Hout, int Wout, int Cout,
    int K, int stride, int pad, int relu) {
  __shared__ __align__(16) _Float16 As[2][64][40];
  __shared__ __align__(16) _Float16 BsT[2][64][40];

  const int tid  = threadIdx.x;
  const int lane = tid & 31;
  const int wave = tid >> 5;
  const int waveM = wave & 3;        // 4 M sub-tiles of 16
  const int waveN = wave >> 2;       // 2 N sub-tiles of 32
  const int hi = (lane >> 4) & 1;
  const int ml = lane & 15;

  const int M0 = blockIdx.x * 64;
  const int N0 = blockIdx.y * 64;
  const int nblk = blockIdx.y;
  const int Mtot = Bn * Hout * Wout;
  const int Kdim = K * K * Cin;
  const int nbN  = Cout >> 6;

  // A staging map: 64 rows x 32 K halfs; 16B per thread, groups in natural order
  const int arow = tid >> 2;
  const int agrp = tid & 3;
  const int apg  = agrp;
  const int am_  = M0 + arow;
  int bb = 0, ho = 0, wo = 0, apix = 0;
  if (am_ < Mtot) {
    wo = am_ % Wout; int t2 = am_ / Wout; ho = t2 % Hout; bb = t2 / Hout; apix = 1;
  }
  // B staging map: straight copy of contiguous 4KB chunk region
  const int bnl = tid >> 2;          // n_local 0..63
  const int bkl = (tid & 3) << 3;    // k_local {0,8,16,24}

  v8f acc0 = {0.f,0.f,0.f,0.f,0.f,0.f,0.f,0.f};
  v8f acc1 = {0.f,0.f,0.f,0.f,0.f,0.f,0.f,0.f};

  const int nch = Kdim >> 5;

  auto stage = [&](int c, int s) {
    const int k0    = c << 5;
    const int tap   = k0 / Cin;
    const int cbase = k0 - tap * Cin;
    const int kh    = tap / K;
    const int kw    = tap - kh * K;
    // A tile
    _Float16* adst = &As[s][arow][apg << 3];
    int hin = ho * stride - pad + kh;
    int win = wo * stride - pad + kw;
    if (apix && hin >= 0 && hin < Hin && win >= 0 && win < Win) {
      const _Float16* asrc =
          X + (size_t)((bb * Hin + hin) * Win + win) * Cin + cbase + (agrp << 3);
      cp16_to_lds(asrc, adst);
    } else {
      uint4 z = make_uint4(0, 0, 0, 0);
      *(uint4*)adst = z;
    }
    // B tile (contiguous per (kChunk, N0) thanks to repack layout)
    const _Float16* bsrc = Wt + ((size_t)c * nbN + nblk) * 2048 + bnl * 32 + bkl;
    cp16_to_lds(bsrc, &BsT[s][bnl][bkl]);
    if (c + 1 < nch)
      __builtin_prefetch(bsrc + (size_t)2048 * nbN, 0, 1);  // global_prefetch_b8
  };

  stage(0, 0);
  async_wait_all();
  __syncthreads();

  for (int c = 0; c < nch; c++) {
    if (c + 1 < nch) stage(c + 1, (c + 1) & 1);
    const int s = c & 1;

    F16Frag af, bf0, bf1;
    const int amr = (waveM << 4) + ml;
    af.h[0] = *(const v8h*)&As[s][amr][(hi << 4)];
    af.h[1] = *(const v8h*)&As[s][amr][(hi << 4) + 8];
    const int nc0 = (waveN << 5) + ml;
    bf0.h[0] = *(const v8h*)&BsT[s][nc0][(hi << 4)];
    bf0.h[1] = *(const v8h*)&BsT[s][nc0][(hi << 4) + 8];
    bf1.h[0] = *(const v8h*)&BsT[s][nc0 + 16][(hi << 4)];
    bf1.h[1] = *(const v8h*)&BsT[s][nc0 + 16][(hi << 4) + 8];

    acc0 = __builtin_amdgcn_wmma_f32_16x16x32_f16(false, af.v, false, bf0.v,
                                                  (short)0, acc0, false, false);
    acc1 = __builtin_amdgcn_wmma_f32_16x16x32_f16(false, af.v, false, bf1.v,
                                                  (short)0, acc1, false, false);
    async_wait_all();
    __syncthreads();
  }

  // epilogue: C layout -> lane = N (mod 16), VGPR r -> M = r + 8*hi
  const int nA = N0 + (waveN << 5) + ml;
  const int nB = nA + 16;
  float scA = bng[nA] * rsqrtf(bnv[nA] + 1e-5f);
  float shA = bnb[nA] - bnm[nA] * scA;
  float scB = bng[nB] * rsqrtf(bnv[nB] + 1e-5f);
  float shB = bnb[nB] - bnm[nB] * scB;
  #pragma unroll
  for (int r2 = 0; r2 < 8; r2++) {
    int mg = M0 + (waveM << 4) + (hi << 3) + r2;
    if (mg < Mtot) {
      float vA = acc0[r2] * scA + shA;
      float vB = acc1[r2] * scB + shB;
      if (Res) {
        vA += (float)Res[(size_t)mg * Cout + nA];
        vB += (float)Res[(size_t)mg * Cout + nB];
      }
      if (relu) { vA = fmaxf(vA, 0.0f); vB = fmaxf(vB, 0.0f); }
      Y[(size_t)mg * Cout + nA] = (_Float16)vA;
      Y[(size_t)mg * Cout + nB] = (_Float16)vB;
    }
  }
}

// ---------------------------------------------------------------------------
// CBAM: channel attention (reduce + 2-layer MLP + sigmoid), apply;
// spatial attention (mean/max over C, 7x7 conv 2ch, sigmoid), apply.
// ---------------------------------------------------------------------------
__global__ __launch_bounds__(256) void ca_kernel(const _Float16* __restrict__ X,
                                                 const float* __restrict__ w1,
                                                 const float* __restrict__ w2,
                                                 float* __restrict__ att,
                                                 int HW, int C, int hidden) {
  __shared__ float avg[512], mxv[512], h1[32], h2[32];
  int b = blockIdx.x;
  const _Float16* xb = X + (size_t)b * HW * C;
  for (int c = threadIdx.x; c < C; c += blockDim.x) {
    float s = 0.0f, m = -1e30f;
    for (int p = 0; p < HW; p++) {
      float vv = (float)xb[(size_t)p * C + c];
      s += vv; m = fmaxf(m, vv);
    }
    avg[c] = s / (float)HW; mxv[c] = m;
  }
  __syncthreads();
  if ((int)threadIdx.x < hidden) {
    float s1 = 0.0f, s2 = 0.0f;
    for (int c = 0; c < C; c++) {
      float w = w1[threadIdx.x * C + c];
      s1 += w * avg[c]; s2 += w * mxv[c];
    }
    h1[threadIdx.x] = fmaxf(s1, 0.0f); h2[threadIdx.x] = fmaxf(s2, 0.0f);
  }
  __syncthreads();
  for (int c = threadIdx.x; c < C; c += blockDim.x) {
    float s = 0.0f;
    for (int h = 0; h < hidden; h++) s += w2[c * hidden + h] * (h1[h] + h2[h]);
    att[b * C + c] = 1.0f / (1.0f + expf(-s));
  }
}

__global__ void ca_apply_kernel(_Float16* X, const float* __restrict__ att,
                                int HWC, int C, int tot) {
  int idx = blockIdx.x * blockDim.x + threadIdx.x;
  if (idx >= tot) return;
  int c = idx % C; int b = idx / HWC;
  X[idx] = (_Float16)((float)X[idx] * att[b * C + c]);
}

__global__ void sa_am_kernel(const _Float16* __restrict__ X, float* __restrict__ am,
                             int totpix, int C) {
  int idx = blockIdx.x * blockDim.x + threadIdx.x;
  if (idx >= totpix) return;
  const _Float16* xp = X + (size_t)idx * C;
  float s = 0.0f, m = -1e30f;
  for (int c = 0; c < C; c++) { float v = (float)xp[c]; s += v; m = fmaxf(m, v); }
  am[(size_t)idx * 2 + 0] = s / (float)C;
  am[(size_t)idx * 2 + 1] = m;
}

__global__ void sa_apply_kernel(_Float16* X, const float* __restrict__ am,
                                const float* __restrict__ saw,
                                int Bn, int H, int W, int C) {
  int idx = blockIdx.x * blockDim.x + threadIdx.x;
  int tot = Bn * H * W;
  if (idx >= tot) return;
  int w = idx % W; int t = idx / W; int h = t % H; int b = t / H;
  float s = 0.0f;
  for (int kh = 0; kh < 7; kh++)
    for (int kw = 0; kw < 7; kw++) {
      int hh = h - 3 + kh, ww = w - 3 + kw;
      if (hh >= 0 && hh < H && ww >= 0 && ww < W) {
        const float* a2 = am + (size_t)((b * H + hh) * W + ww) * 2;
        s += saw[kh * 7 + kw] * a2[0] + saw[49 + kh * 7 + kw] * a2[1];
      }
    }
  float sg = 1.0f / (1.0f + expf(-s));
  _Float16* xp = X + (size_t)idx * C;
  for (int c = 0; c < C; c++) xp[c] = (_Float16)((float)xp[c] * sg);
}

// ---------------------------------------------------------------------------
// ASP attentive stat pooling + BN; FiLM + final FC
// ---------------------------------------------------------------------------
__global__ void asp_mean_kernel(const _Float16* __restrict__ X, float* __restrict__ xm,
                                int Bn, int H, int T, int C) {
  int idx = blockIdx.x * blockDim.x + threadIdx.x;
  int tot = Bn * C * T;
  if (idx >= tot) return;
  int t = idx % T; int r = idx / T; int c = r % C; int b = r / C;
  float s = 0.0f;
  for (int h = 0; h < H; h++) s += (float)X[(size_t)((b * H + h) * T + t) * C + c];
  xm[idx] = s / (float)H;
}

__global__ void asp_h_kernel(const float* __restrict__ xm,
                             const float* __restrict__ w1, const float* __restrict__ b1,
                             const float* g, const float* bb, const float* m, const float* v,
                             float* __restrict__ hb, int Bn, int T) {
  int idx = blockIdx.x * blockDim.x + threadIdx.x;
  int tot = Bn * 128 * T;
  if (idx >= tot) return;
  int t = idx % T; int r = idx / T; int hc = r % 128; int b = r / 128;
  float s = b1[hc];
  for (int c = 0; c < 512; c++) s += w1[hc * 512 + c] * xm[(size_t)(b * 512 + c) * T + t];
  s = fmaxf(s, 0.0f);
  float sc = g[hc] * rsqrtf(v[hc] + 1e-5f);
  hb[(size_t)(b * 128 + hc) * T + t] = (s - m[hc]) * sc + bb[hc];
}

__global__ void asp_logits_kernel(const float* __restrict__ hb,
                                  const float* __restrict__ w2, const float* __restrict__ b2,
                                  float* __restrict__ lg, int Bn, int T) {
  int idx = blockIdx.x * blockDim.x + threadIdx.x;
  int tot = Bn * 512 * T;
  if (idx >= tot) return;
  int t = idx % T; int r = idx / T; int c = r % 512; int b = r / 512;
  float s = b2[c];
  for (int h = 0; h < 128; h++) s += w2[c * 128 + h] * hb[(size_t)(b * 128 + h) * T + t];
  lg[(size_t)(b * 512 + c) * T + t] = s;
}

__global__ void asp_pool_kernel(const float* __restrict__ lg, const float* __restrict__ xm,
                                const float* ag, const float* ab, const float* amn, const float* av,
                                float* __restrict__ feats, int Bn, int T) {
  int idx = blockIdx.x * blockDim.x + threadIdx.x;
  if (idx >= Bn * 512) return;
  int c = idx % 512; int b = idx / 512;
  const float* lr = lg + (size_t)idx * T;
  const float* xr = xm + (size_t)idx * T;
  float mx = -1e30f;
  for (int t = 0; t < T; t++) mx = fmaxf(mx, lr[t]);
  float wv[16]; float ssum = 0.0f;
  for (int t = 0; t < T; t++) { wv[t] = expf(lr[t] - mx); ssum += wv[t]; }
  float mu = 0.0f;
  for (int t = 0; t < T; t++) { wv[t] /= ssum; mu += wv[t] * xr[t]; }
  float var = 0.0f;
  for (int t = 0; t < T; t++) { float d = xr[t] - mu; var += wv[t] * d * d; }
  float sd = sqrtf(var + 1e-6f);
  int f0 = c, f1 = 512 + c;
  feats[b * 1024 + f0] = (mu - amn[f0]) * ag[f0] * rsqrtf(av[f0] + 1e-5f) + ab[f0];
  feats[b * 1024 + f1] = (sd - amn[f1]) * ag[f1] * rsqrtf(av[f1] + 1e-5f) + ab[f1];
}

__global__ __launch_bounds__(256) void film_fc_kernel(
    const float* __restrict__ feats, const int* __restrict__ sal,
    const float* __restrict__ emb,
    const float* __restrict__ w1, const float* __restrict__ b1,
    const float* __restrict__ w2, const float* __restrict__ b2,
    const float* __restrict__ fcw, const float* __restrict__ fcb,
    float* __restrict__ out) {
  __shared__ float sv[64], sh[256], gb[2048], fp[1024];
  int b = blockIdx.x, tid = threadIdx.x;
  int s = sal[b];
  if (tid < 64) sv[tid] = emb[s * 64 + tid];
  __syncthreads();
  {
    float acc = b1[tid];
    for (int k = 0; k < 64; k++) acc += w1[tid * 64 + k] * sv[k];
    sh[tid] = fmaxf(acc, 0.0f);
  }
  __syncthreads();
  for (int j = tid; j < 2048; j += 256) {
    float acc = b2[j];
    for (int k = 0; k < 256; k++) acc += w2[j * 256 + k] * sh[k];
    gb[j] = acc;
  }
  __syncthreads();
  for (int j = tid; j < 1024; j += 256)
    fp[j] = feats[b * 1024 + j] * (1.0f + gb[j]) + gb[1024 + j];
  __syncthreads();
  if (tid < 10) {
    float acc = fcb[tid];
    for (int k = 0; k < 1024; k++) acc += fcw[tid * 1024 + k] * fp[k];
    out[b * 10 + tid] = acc;
  }
}

// ---------------------------------------------------------------------------
// Host orchestration
// ---------------------------------------------------------------------------
extern "C" void kernel_launch(void* const* d_in, const int* in_sizes, int n_in,
                              void* d_out, int out_size, void* d_ws, size_t ws_size,
                              hipStream_t stream) {
  (void)in_sizes; (void)n_in; (void)out_size; (void)ws_size;
  auto F = [&](int i) { return (const float*)d_in[i]; };
  auto gsz = [](long long n, int bsz) { return (unsigned)((n + bsz - 1) / bsz); };
  const int B = 32;

  // ---- input index map (depth-first insertion order of setup_inputs) ----
  const float* mel = F(0);
  int layW1[4][6], layBN1[4][6], layW2[4][6], layBN2[4][6], layDW[4] = {0,0,0,0}, layDBN[4] = {0,0,0,0};
  int cbamW1[4], cbamW2[4], cbamSA[4];
  const int NBv[4] = {3, 4, 6, 3};
  int p = 14;
  for (int li = 0; li < 4; li++) {
    for (int bi = 0; bi < NBv[li]; bi++) {
      layW1[li][bi] = p; layBN1[li][bi] = p + 1;
      layW2[li][bi] = p + 5; layBN2[li][bi] = p + 6; p += 10;
      if (bi == 0 && li > 0) { layDW[li] = p; layDBN[li] = p + 1; p += 5; }
    }
    cbamW1[li] = p; cbamW2[li] = p + 1; cbamSA[li] = p + 2; p += 3;
  }
  int aspW1 = p, aspB1 = p + 1, aspBN = p + 2, aspW2 = p + 6, aspB2 = p + 7; p += 8;
  int aspbn = p; p += 4;
  int embI = p, fw1 = p + 1, fb1 = p + 2, fw2 = p + 3, fb2 = p + 4, fcwI = p + 5, fcbI = p + 6; p += 7;
  const int* sal = (const int*)d_in[p];

  // ---- workspace carve ----
  char* base = (char*)d_ws; size_t off = 0;
  auto alloc = [&](size_t bytes) -> void* {
    void* r = base + off; off = (off + bytes + 255) & ~(size_t)255; return r;
  };
  float*     spec    = (float*)alloc((size_t)B * 128 * 401 * 4);
  _Float16*  actbig  = (_Float16*)alloc((size_t)B * 64 * 201 * 64 * 2);
  size_t     abytes  = (size_t)B * 32 * 101 * 64 * 2;
  _Float16*  buf0    = (_Float16*)alloc(abytes);
  _Float16*  buf1    = (_Float16*)alloc(abytes);
  _Float16*  buf2    = (_Float16*)alloc(abytes);
  _Float16*  wpack   = (_Float16*)alloc((size_t)512 * 512 * 9 * 2);
  float*     att     = (float*)alloc((size_t)B * 512 * 4);
  float*     am      = (float*)alloc((size_t)B * 32 * 101 * 2 * 4);
  float*     xm      = (float*)alloc((size_t)B * 512 * 13 * 4);
  float*     hb      = (float*)alloc((size_t)B * 128 * 13 * 4);
  float*     lg      = (float*)alloc((size_t)B * 512 * 13 * 4);
  float*     feats   = (float*)alloc((size_t)B * 1024 * 4);

  // ---- stem ----
  pcen_kernel<<<gsz(B * 128, 128), 128, 0, stream>>>(
      mel, F(1), F(2), F(3), F(4), F(5), F(6), F(7), F(8), spec, B * 128, 401);
  conv1_kernel<<<gsz((long long)B * 64 * 201, 256), 256, 0, stream>>>(
      spec, F(9), F(10), F(11), F(12), F(13), actbig, B, 128, 401, 64, 201);
  maxpool_kernel<<<gsz((long long)B * 32 * 101 * 64, 256), 256, 0, stream>>>(
      actbig, buf0, B, 64, 201, 32, 101, 64);

  auto run_conv = [&](const _Float16* X, int widx, int bnidx, int Cin, int Cout, int K,
                      int H, int W, int Ho, int Wo, int stride, int pad,
                      const _Float16* res, int relu, _Float16* Y) {
    int total = Cout * Cin * K * K;
    repack_w_kernel<<<gsz(total, 256), 256, 0, stream>>>(F(widx), wpack, Cout, Cin, K, total);
    int M = B * Ho * Wo;
    dim3 grid(gsz(M, 64), Cout / 64);
    conv_wmma_kernel<<<grid, 256, 0, stream>>>(
        X, wpack, F(bnidx), F(bnidx + 1), F(bnidx + 2), F(bnidx + 3),
        res, Y, B, H, W, Cin, Ho, Wo, Cout, K, stride, pad, relu);
  };

  // ---- residual stages ----
  _Float16* x = buf0; _Float16* ta = buf1; _Float16* tb = buf2;
  int H = 32, W = 101, Cin = 64;
  const int CH[4] = {64, 128, 256, 512};
  const int ST[4] = {1, 2, 2, 2};
  for (int li = 0; li < 4; li++) {
    int Cout = CH[li];
    for (int bi = 0; bi < NBv[li]; bi++) {
      int st = (bi == 0) ? ST[li] : 1;
      int Ho = (H + 2 - 3) / st + 1;
      int Wo = (W + 2 - 3) / st + 1;
      bool down = (bi == 0 && li > 0);
      run_conv(x, layW1[li][bi], layBN1[li][bi], Cin, Cout, 3, H, W, Ho, Wo, st, 1,
               nullptr, 1, ta);
      if (down) {
        run_conv(x, layDW[li], layDBN[li], Cin, Cout, 1, H, W, Ho, Wo, st, 0,
                 nullptr, 0, tb);
        run_conv(ta, layW2[li][bi], layBN2[li][bi], Cout, Cout, 3, Ho, Wo, Ho, Wo, 1, 1,
                 tb, 1, x);
      } else {
        run_conv(ta, layW2[li][bi], layBN2[li][bi], Cout, Cout, 3, Ho, Wo, Ho, Wo, 1, 1,
                 x, 1, tb);
        _Float16* t = x; x = tb; tb = t;
      }
      H = Ho; W = Wo; Cin = Cout;
    }
    // CBAM
    int hidden = Cin / 16; if (hidden < 1) hidden = 1;
    int HW = H * W; int tot = B * HW * Cin;
    ca_kernel<<<B, 256, 0, stream>>>(x, F(cbamW1[li]), F(cbamW2[li]), att, HW, Cin, hidden);
    ca_apply_kernel<<<gsz(tot, 256), 256, 0, stream>>>(x, att, HW * Cin, Cin, tot);
    sa_am_kernel<<<gsz(B * HW, 256), 256, 0, stream>>>(x, am, B * HW, Cin);
    sa_apply_kernel<<<gsz(B * HW, 128), 128, 0, stream>>>(x, am, F(cbamSA[li]), B, H, W, Cin);
  }

  // ---- ASP + FiLM + FC ----  (x: [B, H=4, T=13, C=512] NHWC f16)
  asp_mean_kernel<<<gsz((long long)B * 512 * 13, 256), 256, 0, stream>>>(x, xm, B, 4, 13, 512);
  asp_h_kernel<<<gsz((long long)B * 128 * 13, 128), 128, 0, stream>>>(
      xm, F(aspW1), F(aspB1), F(aspBN), F(aspBN + 1), F(aspBN + 2), F(aspBN + 3), hb, B, 13);
  asp_logits_kernel<<<gsz((long long)B * 512 * 13, 128), 128, 0, stream>>>(
      hb, F(aspW2), F(aspB2), lg, B, 13);
  asp_pool_kernel<<<gsz(B * 512, 128), 128, 0, stream>>>(
      lg, xm, F(aspbn), F(aspbn + 1), F(aspbn + 2), F(aspbn + 3), feats, B, 13);
  film_fc_kernel<<<B, 256, 0, stream>>>(
      feats, sal, F(embI), F(fw1), F(fb1), F(fw2), F(fb2), F(fcwI), F(fcbI),
      (float*)d_out);
}
